// MultAttention_36404142801652
// MI455X (gfx1250) — matrix operand order repaired and checked by reading, not compiled, not run on hardware
//
#include <hip/hip_runtime.h>

#define DIMM   768
#define INNER  512
#define NHEADS 8
#define DHEAD  64
#define BATCH  2
#define SEQ    4096
#define NTOK   (BATCH * SEQ)   // 8192

typedef __attribute__((ext_vector_type(16))) __bf16    v16bf;
typedef __attribute__((ext_vector_type(8)))  float     v8f;
typedef __attribute__((ext_vector_type(4)))  uint32_t  u32x4;
typedef __attribute__((ext_vector_type(8)))  uint32_t  u32x8;

// ---------------------------------------------------------------------------
// CDNA5 async / wait primitives
// ---------------------------------------------------------------------------
__device__ inline void async_copy_b128(void* lds_dst, const void* gsrc) {
    // LDS aperture: low 32 bits of a generic LDS pointer are the LDS byte addr.
    uint32_t l = (uint32_t)(uintptr_t)lds_dst;
    asm volatile("global_load_async_to_lds_b128 %0, %1, off"
                 :: "v"(l), "v"(gsrc) : "memory");
}
__device__ inline void wait_asynccnt_0() {
    asm volatile("s_wait_asynccnt 0x0" ::: "memory");
}
__device__ inline void wait_asynccnt_8() {
    asm volatile("s_wait_asynccnt 0x8" ::: "memory");
}
__device__ inline void tdm_load_2d(u32x4 g0, u32x8 g1) {
    asm volatile("tensor_load_to_lds %0, %1" :: "s"(g0), "s"(g1) : "memory");
}

// ---------------------------------------------------------------------------
// WMMA helpers (gfx1250, wave32)
// ---------------------------------------------------------------------------
__device__ inline v8f wmma_bf16(v16bf a, v16bf b, v8f c) {
    return __builtin_amdgcn_wmma_f32_16x16x32_bf16(false, a, false, b,
                                                   (short)0, c, false, false);
}

// A-matrix 16x32 bf16: lanes 0-15: M=lane, K 0..15;
// lanes 16-31: M=lane-16, K 16..31
__device__ inline v16bf load_a_frag(const __bf16* __restrict__ src, int ld) {
    const int lane = threadIdx.x & 31;
    const int m  = lane & 15;
    const int kb = (lane & 16) ? 16 : 0;
    v16bf a;
#pragma unroll
    for (int e = 0; e < 16; ++e) {
        int k = kb + e;
        a[e] = src[(size_t)m * ld + k];
    }
    return a;
}

// B-matrix 32x16 bf16 from row-major [k][n]: lanes 0-15 K=0..15 at N=lane,
// lanes 16-31 K=16..31 at N=lane-16
__device__ inline v16bf load_b_frag(const __bf16* __restrict__ src, int ld) {
    const int lane = threadIdx.x & 31;
    const int n  = lane & 15;
    const int kb = (lane & 16) ? 16 : 0;
    v16bf b;
#pragma unroll
    for (int e = 0; e < 16; ++e)
        b[e] = src[(size_t)(kb + e) * ld + n];
    return b;
}

// B fragment from transposed storage: B[k][n] = src[n*ld + k]
__device__ inline v16bf load_bT_frag(const __bf16* __restrict__ src, int ld) {
    const int lane = threadIdx.x & 31;
    const int n  = lane & 15;
    const int kb = (lane & 16) ? 16 : 0;
    v16bf b;
#pragma unroll
    for (int e = 0; e < 16; ++e)
        b[e] = src[(size_t)n * ld + kb + e];
    return b;
}

__device__ inline float half_reduce_max(float v) {
    v = fmaxf(v, __shfl_xor(v, 1, 32));
    v = fmaxf(v, __shfl_xor(v, 2, 32));
    v = fmaxf(v, __shfl_xor(v, 4, 32));
    v = fmaxf(v, __shfl_xor(v, 8, 32));
    return v;
}
__device__ inline float half_reduce_sum(float v) {
    v += __shfl_xor(v, 1, 32);
    v += __shfl_xor(v, 2, 32);
    v += __shfl_xor(v, 4, 32);
    v += __shfl_xor(v, 8, 32);
    return v;
}

// ---------------------------------------------------------------------------
// fp32 -> bf16 conversion (weights)
// ---------------------------------------------------------------------------
__global__ void cvt_f32_bf16(const float* __restrict__ src,
                             __bf16* __restrict__ dst, int n) {
    int i = blockIdx.x * blockDim.x + threadIdx.x;
    if (i < n) dst[i] = (__bf16)src[i];
}

// ---------------------------------------------------------------------------
// LayerNorm: one 256-thread block per token, bf16 output
// ---------------------------------------------------------------------------
__global__ void __launch_bounds__(256)
layernorm_bf16(const float* __restrict__ x, const float* __restrict__ gamma,
               const float* __restrict__ beta, __bf16* __restrict__ xn) {
    __shared__ float red[256];
    const int t   = blockIdx.x;
    const int tid = threadIdx.x;
    const float* xp = x + (size_t)t * DIMM;

    float s = 0.f;
    for (int i = tid; i < DIMM; i += 256) s += xp[i];
    red[tid] = s; __syncthreads();
    for (int off = 128; off > 0; off >>= 1) {
        if (tid < off) red[tid] += red[tid + off];
        __syncthreads();
    }
    const float mu = red[0] * (1.0f / DIMM);
    __syncthreads();

    float v = 0.f;
    for (int i = tid; i < DIMM; i += 256) { float d = xp[i] - mu; v += d * d; }
    red[tid] = v; __syncthreads();
    for (int off = 128; off > 0; off >>= 1) {
        if (tid < off) red[tid] += red[tid + off];
        __syncthreads();
    }
    const float rs = rsqrtf(red[0] * (1.0f / DIMM) + 1e-5f);

    __bf16* op = xn + (size_t)t * DIMM;
    for (int i = tid; i < DIMM; i += 256)
        op[i] = (__bf16)((xp[i] - mu) * rs * gamma[i] + beta[i]);
}

// ---------------------------------------------------------------------------
// Tiled bf16 WMMA GEMM: C[M,N] = A[M,K] * B[K,N], fp32 accumulate.
// 128 threads = 4 waves; block tile 64x64, K step 32; wave tile 32x32 (2x2 WMMA).
// B tile staged via Tensor Data Mover (wave 0 issues a 2D D#, TENSORcnt);
// A tile staged via GLOBAL_LOAD_ASYNC_TO_LDS_B128 (ASYNCcnt).
// EPI 0: fp32 row-major; EPI 1: Q head layout; EPI 2: KV split head layout.
// ---------------------------------------------------------------------------
template <int EPI>
__global__ void __launch_bounds__(128)
gemm_bf16_wmma(const __bf16* __restrict__ A, const __bf16* __restrict__ Bw,
               void* __restrict__ C0, void* __restrict__ C1,
               int M, int N, int K) {
    __shared__ alignas(16) __bf16 As[64 * 32];
    __shared__ alignas(16) __bf16 Bs[32 * 64];

    const int tid  = threadIdx.x;
    const int wave = tid >> 5;
    const int lane = tid & 31;
    const int cn   = lane & 15;
    const int half = lane >> 4;
    const int m0   = blockIdx.y * 64;
    const int n0   = blockIdx.x * 64;
    const int wm   = (wave >> 1) * 32;
    const int wn   = (wave & 1) * 32;

    // TDM D# group1 (loop-invariant): data_size=2B, tile 64(dim0) x 32(dim1),
    // tensor_dim0 = N, tensor_dim0_stride = N, no padding / iterate / multicast.
    u32x8 g1;
    g1[0] = 1u << 16;                                          // data_size = 2B
    g1[1] = ((uint32_t)N & 0xFFFFu) << 16;                     // tensor_dim0 lo
    g1[2] = ((uint32_t)N >> 16) | (((uint32_t)K & 0xFFFFu) << 16); // dim0 hi | dim1 lo
    g1[3] = (((uint32_t)K >> 16) & 0xFFFFu) | (64u << 16);     // dim1 hi | tile_dim0
    g1[4] = 32u;                                               // tile_dim1 (tile_dim2=0)
    g1[5] = (uint32_t)N;                                       // tensor_dim0_stride lo
    g1[6] = 0u;
    g1[7] = 0u;
    const uint32_t bs_lds = (uint32_t)(uintptr_t)&Bs[0];

    const v8f z = {0.f,0.f,0.f,0.f,0.f,0.f,0.f,0.f};
    v8f c00 = z, c01 = z, c10 = z, c11 = z;

    const int ar = tid >> 1, ac = (tid & 1) * 16;   // A-tile staging coords

    for (int k0 = 0; k0 < K; k0 += 32) {
        if (wave == 0) {   // one TDM 2D tile load for B (uniform branch)
            uint64_t ga = (uint64_t)(uintptr_t)(Bw + (size_t)k0 * N + n0);
            u32x4 g0;
            g0[0] = 1u;                         // count=1, user mode
            g0[1] = bs_lds;                     // lds_addr
            g0[2] = (uint32_t)ga;               // global_addr[31:0]
            g0[3] = (uint32_t)((ga >> 32) & 0x1FFFFFFu) | (2u << 30); // [56:32]|type=2
            tdm_load_2d(g0, g1);
        }
        { // A tile 64x32 via async-to-LDS (2 x b128 per thread)
            const __bf16* g = A + (size_t)(m0 + ar) * K + k0 + ac;
            __bf16* sm = As + ar * 32 + ac;
            async_copy_b128(sm,     g);
            async_copy_b128(sm + 8, g + 8);
        }
        wait_asynccnt_0();
        if (wave == 0) __builtin_amdgcn_s_wait_tensorcnt((short)0);
        __syncthreads();

        v16bf a0 = load_a_frag(As + (size_t)wm * 32, 32);
        v16bf a1 = load_a_frag(As + (size_t)(wm + 16) * 32, 32);
        v16bf b0 = load_b_frag(Bs + wn, 64);
        v16bf b1 = load_b_frag(Bs + wn + 16, 64);
        c00 = wmma_bf16(a0, b0, c00);
        c01 = wmma_bf16(a0, b1, c01);
        c10 = wmma_bf16(a1, b0, c10);
        c11 = wmma_bf16(a1, b1, c11);
        __syncthreads();
    }

    const v8f* cc[4] = {&c00, &c01, &c10, &c11};
#pragma unroll
    for (int ti = 0; ti < 2; ++ti) {
#pragma unroll
        for (int tj = 0; tj < 2; ++tj) {
            const v8f cv = *cc[ti * 2 + tj];
#pragma unroll
            for (int r = 0; r < 8; ++r) {
                const int row = m0 + wm + ti * 16 + half * 8 + r;
                const int col = n0 + wn + tj * 16 + cn;
                const float val = cv[r];
                if constexpr (EPI == 0) {
                    ((float*)C0)[(size_t)row * N + col] = val;
                } else if constexpr (EPI == 1) {
                    int b = row / SEQ, l = row % SEQ, h = col >> 6, d = col & 63;
                    ((__bf16*)C0)[(((size_t)(b * NHEADS + h)) * SEQ + l) * DHEAD + d] =
                        (__bf16)val;
                } else {
                    int b = row / SEQ, l = row % SEQ;
                    int cc2 = (col < INNER) ? col : (col - INNER);
                    int h = cc2 >> 6, d = cc2 & 63;
                    __bf16* dst = (col < INNER) ? (__bf16*)C0 : (__bf16*)C1;
                    dst[(((size_t)(b * NHEADS + h)) * SEQ + l) * DHEAD + d] = (__bf16)val;
                }
            }
        }
    }
}

// ---------------------------------------------------------------------------
// Flash attention. grid = (B*H, SEQ/64), 128 threads (4 waves).
// Wave owns 16 query rows; Q in A-fragments; K/V 64x64 tiles DOUBLE-BUFFERED in
// LDS via GLOBAL_LOAD_ASYNC_TO_LDS_B128 (8 async b128 per thread per tile;
// s_wait_asynccnt <=8 overlaps next-tile loads with current-tile compute).
// Scores scaled 1/8, fp32 online softmax, PV via WMMA.
// ---------------------------------------------------------------------------
__global__ void __launch_bounds__(128)
flash_attn_wmma(const __bf16* __restrict__ Q, const __bf16* __restrict__ Kg,
                const __bf16* __restrict__ Vg, __bf16* __restrict__ O) {
    __shared__ alignas(16) __bf16 Ks[2][64 * 64];
    __shared__ alignas(16) __bf16 Vs[2][64 * 64];
    __shared__ alignas(16) __bf16 Ps[64 * 64];

    const int tid  = threadIdx.x;
    const int wave = tid >> 5;
    const int lane = tid & 31;
    const int cn   = lane & 15;
    const int half = lane >> 4;
    const int bh   = blockIdx.x;
    const int qrow = blockIdx.y * 64 + wave * 16;

    const __bf16* Qp = Q  + (size_t)bh * SEQ * DHEAD;
    const __bf16* Kp = Kg + (size_t)bh * SEQ * DHEAD;
    const __bf16* Vp = Vg + (size_t)bh * SEQ * DHEAD;

    v16bf qa0 = load_a_frag(Qp + (size_t)qrow * DHEAD, DHEAD);
    v16bf qa1 = load_a_frag(Qp + (size_t)qrow * DHEAD + 32, DHEAD);

    const v8f z = {0.f,0.f,0.f,0.f,0.f,0.f,0.f,0.f};
    v8f o0 = z, o1 = z, o2 = z, o3 = z;
    float m_i[8], l_i[8];
#pragma unroll
    for (int r = 0; r < 8; ++r) { m_i[r] = -1e30f; l_i[r] = 0.f; }

    __bf16* Pw = Ps + (size_t)wave * 16 * 64;

    const int sr = tid >> 1, sc = (tid & 1) * 32;   // staging coords
    auto issue_tile = [&](int buf, int kb0) {
        const __bf16* gk = Kp + (size_t)(kb0 + sr) * DHEAD + sc;
        const __bf16* gv = Vp + (size_t)(kb0 + sr) * DHEAD + sc;
        __bf16* sk = &Ks[buf][sr * 64 + sc];
        __bf16* sv = &Vs[buf][sr * 64 + sc];
#pragma unroll
        for (int u = 0; u < 4; ++u) {
            async_copy_b128(sk + u * 8, gk + u * 8);
            async_copy_b128(sv + u * 8, gv + u * 8);
        }
    };

    issue_tile(0, 0);
    int buf = 0;
    for (int kb0 = 0; kb0 < SEQ; kb0 += 64, buf ^= 1) {
        const bool has_next = (kb0 + 64) < SEQ;
        if (has_next) issue_tile(buf ^ 1, kb0 + 64);   // prefetch next tile
        if (has_next) wait_asynccnt_8();               // current 8 ops done
        else          wait_asynccnt_0();
        __syncthreads();

        const __bf16* Kt = Ks[buf];
        const __bf16* Vt = Vs[buf];

        // S = (Q * K^T) / 8 : four 16x16 tiles, two k-chunks each
        v8f s[4];
#pragma unroll
        for (int nt = 0; nt < 4; ++nt) {
            v8f acc = z;
            acc = wmma_bf16(qa0, load_bT_frag(Kt + (size_t)nt * 16 * 64, 64), acc);
            acc = wmma_bf16(qa1, load_bT_frag(Kt + (size_t)nt * 16 * 64 + 32, 64), acc);
#pragma unroll
            for (int r = 0; r < 8; ++r) acc[r] *= 0.125f;
            s[nt] = acc;
        }

        // online softmax per row (row r+half*8 across the 16 lanes of a half)
        float alpha[8];
#pragma unroll
        for (int r = 0; r < 8; ++r) {
            float mx = fmaxf(fmaxf(s[0][r], s[1][r]), fmaxf(s[2][r], s[3][r]));
            mx = half_reduce_max(mx);
            const float mnew = fmaxf(m_i[r], mx);
            alpha[r] = __expf(m_i[r] - mnew);
            float ls = 0.f;
#pragma unroll
            for (int nt = 0; nt < 4; ++nt) {
                float p = __expf(s[nt][r] - mnew);
                s[nt][r] = p;
                ls += p;
            }
            ls = half_reduce_sum(ls);
            l_i[r] = l_i[r] * alpha[r] + ls;
            m_i[r] = mnew;
        }

        // stage P (bf16) into this wave's private LDS region, reload as A-frags
#pragma unroll
        for (int nt = 0; nt < 4; ++nt)
#pragma unroll
            for (int r = 0; r < 8; ++r)
                Pw[(half * 8 + r) * 64 + nt * 16 + cn] = (__bf16)s[nt][r];
        asm volatile("s_wait_dscnt 0x0" ::: "memory");   // same-wave LDS RAW fence

        v16bf pa0 = load_a_frag(Pw, 64);
        v16bf pa1 = load_a_frag(Pw + 32, 64);

#pragma unroll
        for (int r = 0; r < 8; ++r) {
            o0[r] *= alpha[r]; o1[r] *= alpha[r];
            o2[r] *= alpha[r]; o3[r] *= alpha[r];
        }
        o0 = wmma_bf16(pa0, load_b_frag(Vt + 0,  64), o0);
        o0 = wmma_bf16(pa1, load_b_frag(Vt + 32 * 64 + 0,  64), o0);
        o1 = wmma_bf16(pa0, load_b_frag(Vt + 16, 64), o1);
        o1 = wmma_bf16(pa1, load_b_frag(Vt + 32 * 64 + 16, 64), o1);
        o2 = wmma_bf16(pa0, load_b_frag(Vt + 32, 64), o2);
        o2 = wmma_bf16(pa1, load_b_frag(Vt + 32 * 64 + 32, 64), o2);
        o3 = wmma_bf16(pa0, load_b_frag(Vt + 48, 64), o3);
        o3 = wmma_bf16(pa1, load_b_frag(Vt + 32 * 64 + 48, 64), o3);
        __syncthreads();
    }

    const int b = bh >> 3, h = bh & 7;
#pragma unroll
    for (int r = 0; r < 8; ++r) {
        const float inv = 1.0f / l_i[r];
        const int row = qrow + half * 8 + r;
        const size_t base = ((size_t)(b * SEQ + row)) * INNER + h * DHEAD;
        O[base + 0  + cn] = (__bf16)(o0[r] * inv);
        O[base + 16 + cn] = (__bf16)(o1[r] * inv);
        O[base + 32 + cn] = (__bf16)(o2[r] * inv);
        O[base + 48 + cn] = (__bf16)(o3[r] * inv);
    }
}

// ---------------------------------------------------------------------------
// launch
// ---------------------------------------------------------------------------
extern "C" void kernel_launch(void* const* d_in, const int* in_sizes, int n_in,
                              void* d_out, int out_size, void* d_ws, size_t ws_size,
                              hipStream_t stream) {
    const float* x     = (const float*)d_in[0];
    const float* gamma = (const float*)d_in[1];
    const float* beta  = (const float*)d_in[2];
    const float* Wq    = (const float*)d_in[3];
    const float* Wkv   = (const float*)d_in[4];
    const float* Wo    = (const float*)d_in[5];
    float* out = (float*)d_out;

    char* ws = (char*)d_ws;
    size_t off = 0;
    auto salloc = [&](size_t bytes) -> char* {
        char* p = ws + off;
        off += (bytes + 255) & ~(size_t)255;
        return p;
    };
    __bf16* xn   = (__bf16*)salloc((size_t)NTOK * DIMM * 2);
    __bf16* wqb  = (__bf16*)salloc((size_t)DIMM * INNER * 2);
    __bf16* wkvb = (__bf16*)salloc((size_t)DIMM * 2 * INNER * 2);
    __bf16* wob  = (__bf16*)salloc((size_t)INNER * DIMM * 2);
    __bf16* qb   = (__bf16*)salloc((size_t)NTOK * INNER * 2);
    __bf16* kb   = (__bf16*)salloc((size_t)NTOK * INNER * 2);
    __bf16* vb   = (__bf16*)salloc((size_t)NTOK * INNER * 2);
    __bf16* ab   = (__bf16*)salloc((size_t)NTOK * INNER * 2);

    const int nq  = DIMM * INNER;
    const int nkv = DIMM * 2 * INNER;
    const int no  = INNER * DIMM;
    cvt_f32_bf16<<<dim3((nq  + 255) / 256), 256, 0, stream>>>(Wq,  wqb,  nq);
    cvt_f32_bf16<<<dim3((nkv + 255) / 256), 256, 0, stream>>>(Wkv, wkvb, nkv);
    cvt_f32_bf16<<<dim3((no  + 255) / 256), 256, 0, stream>>>(Wo,  wob,  no);

    layernorm_bf16<<<dim3(NTOK), 256, 0, stream>>>(x, gamma, beta, xn);

    gemm_bf16_wmma<1><<<dim3(INNER / 64, NTOK / 64), 128, 0, stream>>>(
        xn, wqb, (void*)qb, nullptr, NTOK, INNER, DIMM);
    gemm_bf16_wmma<2><<<dim3((2 * INNER) / 64, NTOK / 64), 128, 0, stream>>>(
        xn, wkvb, (void*)kb, (void*)vb, NTOK, 2 * INNER, DIMM);

    flash_attn_wmma<<<dim3(BATCH * NHEADS, SEQ / 64), 128, 0, stream>>>(qb, kb, vb, ab);

    gemm_bf16_wmma<0><<<dim3(DIMM / 64, NTOK / 64), 128, 0, stream>>>(
        ab, wob, (void*)out, nullptr, NTOK, DIMM, INNER);
}
